// Model_81475529605800
// MI455X (gfx1250) — matrix operand order repaired and checked by reading, not compiled, not run on hardware
//
#include <hip/hip_runtime.h>
#include <math.h>

typedef __attribute__((ext_vector_type(16))) _Float16 v16h;
typedef __attribute__((ext_vector_type(8)))  float    v8f;

// ---------------- CDNA5 WMMA helpers (wave32, 16x16x32 f16 -> f32) ----------

__device__ __forceinline__ v8f wmma16(v16h a, v16h b, v8f c) {
  // 8 args: (neg_a, A, neg_b, B, c_mod, C, reuse_a, reuse_b)
  return __builtin_amdgcn_wmma_f32_16x16x32_f16(false, a, false, b, (short)0, c,
                                                false, false);
}

// Same-wave LDS RAW/WAR ordering (DS ops are in-order per wave; drain DScnt)
__device__ __forceinline__ void lds_fence() {
  asm volatile("s_wait_dscnt 0" ::: "memory");
}

// Hardware f32 atomic add (no return -> STOREcnt; s_endpgm drains it)
__device__ __forceinline__ void global_fadd(float* p, float v) {
  asm volatile("global_atomic_add_f32 %0, %1, off" :: "v"(p), "v"(v) : "memory");
}

// A-fragment (16x32 f16), lane = row (lane&15). ISA layout: halves 0..7 hold
// K = base..base+7, halves 8..15 hold K = base+16..base+23, base = (lane>=16)?8:0.
// Full 32-wide f32 row: 4x float4 loads (global_load_b128), branch-free.
__device__ __forceinline__ v16h a_frag_row32(const float* __restrict__ row) {
  const int lane = threadIdx.x & 31;
  const int q = (lane >> 4) << 1;  // float4 index 0 (grp0) or 2 (grp1)
  const float4* p = (const float4*)row;
  const float4 x0 = p[q + 0];      // K base..base+3
  const float4 x1 = p[q + 1];      // K base+4..base+7
  const float4 x2 = p[q + 4];      // K base+16..base+19
  const float4 x3 = p[q + 5];      // K base+20..base+23
  v16h a;
  a[0]  = (_Float16)x0.x; a[1]  = (_Float16)x0.y; a[2]  = (_Float16)x0.z; a[3]  = (_Float16)x0.w;
  a[4]  = (_Float16)x1.x; a[5]  = (_Float16)x1.y; a[6]  = (_Float16)x1.z; a[7]  = (_Float16)x1.w;
  a[8]  = (_Float16)x2.x; a[9]  = (_Float16)x2.y; a[10] = (_Float16)x2.z; a[11] = (_Float16)x2.w;
  a[12] = (_Float16)x3.x; a[13] = (_Float16)x3.y; a[14] = (_Float16)x3.z; a[15] = (_Float16)x3.w;
  return a;
}

// 8-wide f32 row (e_vals), zero-padded to K=32. All lanes issue the (valid)
// loads; lanes 16..31 and halves 8..15 are masked to zero via VALU select —
// no exec-mask branching, no serialized waits.
__device__ __forceinline__ v16h a_frag_row8(const float* __restrict__ row) {
  const int lane = threadIdx.x & 31;
  const float m = (lane < 16) ? 1.0f : 0.0f;  // grp0 covers K 0..7; grp1 K 8..15 (all pad)
  const float4* p = (const float4*)row;
  const float4 x0 = p[0];
  const float4 x1 = p[1];
  v16h a;
  a[0] = (_Float16)(x0.x * m); a[1] = (_Float16)(x0.y * m);
  a[2] = (_Float16)(x0.z * m); a[3] = (_Float16)(x0.w * m);
  a[4] = (_Float16)(x1.x * m); a[5] = (_Float16)(x1.y * m);
  a[6] = (_Float16)(x1.z * m); a[7] = (_Float16)(x1.w * m);
#pragma unroll
  for (int i = 8; i < 16; ++i) a[i] = (_Float16)0.0f;  // K base+16..+23 >= 8
  return a;
}

// A-fragment from an f16 LDS row (full 32-wide chunk starting at k0).
__device__ __forceinline__ v16h a_frag_ldsrow(const _Float16* row, int k0) {
  const int lane = threadIdx.x & 31;
  const int base = k0 + ((lane >> 4) << 3);
  v16h a;
#pragma unroll
  for (int i = 0; i < 8; ++i) {
    a[i]     = row[base + i];
    a[i + 8] = row[base + 16 + i];
  }
  return a;
}

// B-fragment (32x16 f16): lane = column (lane&15); halves i hold
// K = k0 + 8*(lane>=16) + (i&7) + 16*(i>=8). Unconditional clamped load + select (no branches).
__device__ __forceinline__ v16h b_frag(const float* __restrict__ W, int dIn, int dOut,
                                       int k0, int n0) {
  const int lane = threadIdx.x & 31;
  const int col  = n0 + (lane & 15);
  const int kb   = k0 + ((lane >> 4) << 3);
  v16h b;
#pragma unroll
  for (int i = 0; i < 16; ++i) {
    const int k  = kb + (i & 7) + ((i >> 3) << 4);
    const int kc = (k < dIn) ? k : (dIn - 1);
    const float w = W[kc * dOut + col];
    b[i] = (k < dIn) ? (_Float16)w : (_Float16)0.0f;
  }
  return b;
}

// ---------------- kernel 0: zero the aggregation buffer ---------------------

__global__ void zero_f32(float* __restrict__ p, int n) {
  const int i = blockIdx.x * blockDim.x + threadIdx.x;
  if (i < n) p[i] = 0.0f;
}

// ---------------- kernel 1: edge MLP + atomic scatter-add -------------------
// Per 16-edge tile (one wave): edge_in(16x72) @ g_w1(72x64) -> ReLU
//                              -> (16x64) @ g_w2(64x32) -> ReLU -> atomicAdd agg
// K padded 72->96 (3 chunks of 32): [u | v | e_vals(8)+zeros].

#define H1S 68  // LDS row stride in halves (padded vs 64 to avoid bank conflicts)

__global__ void __launch_bounds__(256)
edge_mlp_scatter(const float* __restrict__ u, const float* __restrict__ v,
                 const float* __restrict__ e_vals,
                 const int* __restrict__ e_src, const int* __restrict__ e_dst,
                 const float* __restrict__ g_w1, const float* __restrict__ g_b1,
                 const float* __restrict__ g_w2, const float* __restrict__ g_b2,
                 float* __restrict__ agg, int n_edges, int n_tiles) {
  __shared__ _Float16 lds_h1[8][16 * H1S];  // per-wave layer-1 activations

  const int lane = threadIdx.x & 31;
  const int wid  = threadIdx.x >> 5;
  const int n    = lane & 15;
  const int grp  = lane >> 4;

  // Weight fragments live in registers for the wave's whole lifetime.
  v16h B1[3][4];
#pragma unroll
  for (int c = 0; c < 3; ++c)
#pragma unroll
    for (int nt = 0; nt < 4; ++nt) B1[c][nt] = b_frag(g_w1, 72, 64, c * 32, nt * 16);
  v16h B2[2][2];
#pragma unroll
  for (int c = 0; c < 2; ++c)
#pragma unroll
    for (int nt = 0; nt < 2; ++nt) B2[c][nt] = b_frag(g_w2, 64, 32, c * 32, nt * 16);

  float bias1[4], bias2[2];
#pragma unroll
  for (int nt = 0; nt < 4; ++nt) bias1[nt] = g_b1[nt * 16 + n];
#pragma unroll
  for (int nt = 0; nt < 2; ++nt) bias2[nt] = g_b2[nt * 16 + n];

  _Float16* hrow = lds_h1[wid];
  const int wave   = blockIdx.x * 8 + wid;
  const int stride = gridDim.x * 8;

  for (int t = wave; t < n_tiles; t += stride) {
    const int tbase = t * 16;
    int erow = tbase + n;
    if (erow >= n_edges) erow = n_edges - 1;  // duplicate row; scatter is guarded
    const int iu = e_dst[erow];
    const int iv = e_src[erow];

    // ---- layer 1: 12 WMMAs ----
    v8f acc[4];
#pragma unroll
    for (int nt = 0; nt < 4; ++nt)
#pragma unroll
      for (int j = 0; j < 8; ++j) acc[nt][j] = bias1[nt];

    {
      v16h A = a_frag_row32(u + (size_t)iu * 32);          // K 0..31 (u gather, L2)
#pragma unroll
      for (int nt = 0; nt < 4; ++nt) acc[nt] = wmma16(A, B1[0][nt], acc[nt]);
    }
    {
      v16h A = a_frag_row32(v + (size_t)iv * 32);          // K 32..63 (v gather, L2)
#pragma unroll
      for (int nt = 0; nt < 4; ++nt) acc[nt] = wmma16(A, B1[1][nt], acc[nt]);
    }
    {
      v16h A = a_frag_row8(e_vals + (size_t)erow * 8);     // K 64..71 + zero pad
#pragma unroll
      for (int nt = 0; nt < 4; ++nt) acc[nt] = wmma16(A, B1[2][nt], acc[nt]);
    }

    // ReLU, f16, redistribute C-layout -> row-major LDS for layer-2 A fragments
#pragma unroll
    for (int nt = 0; nt < 4; ++nt) {
      const int col = nt * 16 + n;
#pragma unroll
      for (int j = 0; j < 8; ++j)
        hrow[(grp * 8 + j) * H1S + col] = (_Float16)fmaxf(acc[nt][j], 0.0f);
    }
    lds_fence();

    // ---- layer 2: 4 WMMAs ----
    v8f acc2[2];
#pragma unroll
    for (int nt = 0; nt < 2; ++nt)
#pragma unroll
      for (int j = 0; j < 8; ++j) acc2[nt][j] = bias2[nt];
#pragma unroll
    for (int c = 0; c < 2; ++c) {
      v16h A2 = a_frag_ldsrow(hrow + n * H1S, c * 32);
#pragma unroll
      for (int nt = 0; nt < 2; ++nt) acc2[nt] = wmma16(A2, B2[c][nt], acc2[nt]);
    }
    lds_fence();  // keep next iteration's LDS stores behind this tile's reads

    // ---- ReLU + scatter-add into agg[U][32] ----
    // lane n already holds e_dst for row n; broadcast via wave shuffle instead
    // of re-reading global memory.
#pragma unroll
    for (int j = 0; j < 8; ++j) {
      const int mr  = grp * 8 + j;
      const int idx = __shfl(iu, mr, 32);
      const int row = tbase + mr;
      if (row < n_edges) {
        float* pbase = agg + (size_t)idx * 32;
        global_fadd(pbase + n,      fmaxf(acc2[0][j], 0.0f));
        global_fadd(pbase + 16 + n, fmaxf(acc2[1][j], 0.0f));
      }
    }
  }
}

// ---------------- kernel 2: node MLP + tail + sigmoid -----------------------
// Per 16-node tile: [u | agg](16x64) @ f_w1(64x64) -> ReLU
//                   -> @ f_w2(64x32) -> ReLU -> @ t_w(32x1) + t_b -> sigmoid

#define H2S 33  // LDS f32 row stride (padded vs 32 to avoid bank conflicts)

__global__ void __launch_bounds__(256)
node_mlp(const float* __restrict__ u, const float* __restrict__ agg,
         const float* __restrict__ f_w1, const float* __restrict__ f_b1,
         const float* __restrict__ f_w2, const float* __restrict__ f_b2,
         const float* __restrict__ t_w, const float* __restrict__ t_b,
         float* __restrict__ out, int n_nodes, int n_tiles) {
  __shared__ _Float16 lds_z[8][16 * H1S];
  __shared__ float    lds_h[8][16 * H2S];

  const int lane = threadIdx.x & 31;
  const int wid  = threadIdx.x >> 5;
  const int n    = lane & 15;
  const int grp  = lane >> 4;

  v16h B1[2][4];
#pragma unroll
  for (int c = 0; c < 2; ++c)
#pragma unroll
    for (int nt = 0; nt < 4; ++nt) B1[c][nt] = b_frag(f_w1, 64, 64, c * 32, nt * 16);
  v16h B2[2][2];
#pragma unroll
  for (int c = 0; c < 2; ++c)
#pragma unroll
    for (int nt = 0; nt < 2; ++nt) B2[c][nt] = b_frag(f_w2, 64, 32, c * 32, nt * 16);

  float bias1[4], bias2[2];
#pragma unroll
  for (int nt = 0; nt < 4; ++nt) bias1[nt] = f_b1[nt * 16 + n];
#pragma unroll
  for (int nt = 0; nt < 2; ++nt) bias2[nt] = f_b2[nt * 16 + n];
  const float tb = t_b[0];

  float tw[32];  // tail weights resident in registers
#pragma unroll
  for (int k = 0; k < 32; ++k) tw[k] = t_w[k];

  _Float16* zrow = lds_z[wid];
  float*    hrw  = lds_h[wid];
  const int wave   = blockIdx.x * 8 + wid;
  const int stride = gridDim.x * 8;

  for (int t = wave; t < n_tiles; t += stride) {
    const int tbase = t * 16;
    int row = tbase + n;
    if (row >= n_nodes) row = n_nodes - 1;

    v8f acc[4];
#pragma unroll
    for (int nt = 0; nt < 4; ++nt)
#pragma unroll
      for (int j = 0; j < 8; ++j) acc[nt][j] = bias1[nt];

    {
      v16h A = a_frag_row32(u + (size_t)row * 32);    // K 0..31  = u
#pragma unroll
      for (int nt = 0; nt < 4; ++nt) acc[nt] = wmma16(A, B1[0][nt], acc[nt]);
    }
    {
      v16h A = a_frag_row32(agg + (size_t)row * 32);  // K 32..63 = agg
#pragma unroll
      for (int nt = 0; nt < 4; ++nt) acc[nt] = wmma16(A, B1[1][nt], acc[nt]);
    }

#pragma unroll
    for (int nt = 0; nt < 4; ++nt) {
      const int col = nt * 16 + n;
#pragma unroll
      for (int j = 0; j < 8; ++j)
        zrow[(grp * 8 + j) * H1S + col] = (_Float16)fmaxf(acc[nt][j], 0.0f);
    }
    lds_fence();

    v8f acc2[2];
#pragma unroll
    for (int nt = 0; nt < 2; ++nt)
#pragma unroll
      for (int j = 0; j < 8; ++j) acc2[nt][j] = bias2[nt];
#pragma unroll
    for (int c = 0; c < 2; ++c) {
      v16h A2 = a_frag_ldsrow(zrow + n * H1S, c * 32);
#pragma unroll
      for (int nt = 0; nt < 2; ++nt) acc2[nt] = wmma16(A2, B2[c][nt], acc2[nt]);
    }

#pragma unroll
    for (int nt = 0; nt < 2; ++nt) {
      const int col = nt * 16 + n;
#pragma unroll
      for (int j = 0; j < 8; ++j)
        hrw[(grp * 8 + j) * H2S + col] = fmaxf(acc2[nt][j], 0.0f);
    }
    lds_fence();

    // tail: 32-wide dot + bias + sigmoid, one lane per row
    if (lane < 16 && (tbase + lane) < n_nodes) {
      float s = tb;
#pragma unroll
      for (int k = 0; k < 32; ++k) s += hrw[lane * H2S + k] * tw[k];
      out[tbase + lane] = 1.0f / (1.0f + expf(-s));
    }
    lds_fence();  // next iteration overwrites zrow/hrw
  }
}

// ---------------- host-side launcher ---------------------------------------

extern "C" void kernel_launch(void* const* d_in, const int* in_sizes, int n_in,
                              void* d_out, int out_size, void* d_ws, size_t ws_size,
                              hipStream_t stream) {
  const float* u      = (const float*)d_in[0];
  const float* v      = (const float*)d_in[1];
  const float* e_vals = (const float*)d_in[2];
  const int*   e_src  = (const int*)d_in[3];
  const int*   e_dst  = (const int*)d_in[4];
  const float* g_w1   = (const float*)d_in[5];
  const float* g_b1   = (const float*)d_in[6];
  const float* g_w2   = (const float*)d_in[7];
  const float* g_b2   = (const float*)d_in[8];
  const float* f_w1   = (const float*)d_in[9];
  const float* f_b1   = (const float*)d_in[10];
  const float* f_w2   = (const float*)d_in[11];
  const float* f_b2   = (const float*)d_in[12];
  const float* t_w    = (const float*)d_in[13];
  const float* t_b    = (const float*)d_in[14];

  const int n_nodes = in_sizes[0] / 32;  // U
  const int n_edges = in_sizes[3];       // E
  float* agg = (float*)d_ws;             // U x 32 f32 scratch (12.8 MB, L2-resident)
  float* out = (float*)d_out;

  const int agg_elems = n_nodes * 32;
  zero_f32<<<(agg_elems + 255) / 256, 256, 0, stream>>>(agg, agg_elems);

  const int e_tiles = (n_edges + 15) / 16;      // 125,000
  int eblocks = (e_tiles + 7) / 8;
  if (eblocks > 1536) eblocks = 1536;           // 12,288 waves, ~10 tiles each
  edge_mlp_scatter<<<eblocks, 256, 0, stream>>>(u, v, e_vals, e_src, e_dst,
                                                g_w1, g_b1, g_w2, g_b2,
                                                agg, n_edges, e_tiles);

  const int u_tiles = (n_nodes + 15) / 16;      // 6,250
  int ublocks = (u_tiles + 7) / 8;
  if (ublocks > 512) ublocks = 512;
  node_mlp<<<ublocks, 256, 0, stream>>>(u, agg, f_w1, f_b1, f_w2, f_b2,
                                        t_w, t_b, out, n_nodes, u_tiles);
}
